// PatientNodeClassifier_42082089566423
// MI455X (gfx1250) — matrix operand order repaired and checked by reading, not compiled, not run on hardware
//
#include <hip/hip_runtime.h>
#include <math.h>

typedef __attribute__((ext_vector_type(16))) __bf16 v16bf;
typedef __attribute__((ext_vector_type(8)))  float  v8f;

#define DIMF 128
#define HDS  4
#define CHN  32

__device__ __forceinline__ float4 ld4(const float* p){ return *(const float4*)p; }

// ---------------- prep: embedding bias sum, zero BN accumulators ----------------
__global__ void k_prep(const float* __restrict__ emb_b, float* bsum, float* bns, float* bnq){
  int c = threadIdx.x;
  if (c < DIMF){
    bsum[c] = emb_b[c] + emb_b[DIMF + c] + emb_b[2*DIMF + c];
    bns[c] = 0.f; bnq[c] = 0.f;
  }
}

// ---------------- pack weight B[K,128] fp32 -> fragment-ordered bf16 ----------------
// P linear layout: (((ct*nsteps + st)*32 + lane)*16 + e)
//   k = st*32 + ((lane&16)>>1) + (e&7) + ((e&8)<<1) ; col = ct*16 + (lane&15) ; zero-padded past K
__global__ void k_packB(const float* __restrict__ W, __bf16* __restrict__ P, int K){
  int nsteps = (K + 31) / 32;
  int total  = 8 * nsteps * 32 * 16;
  int idx = blockIdx.x * blockDim.x + threadIdx.x;
  if (idx >= total) return;
  int e    = idx & 15;
  int lane = (idx >> 4) & 31;
  int rest = idx >> 9;              // ct*nsteps + st
  int st   = rest % nsteps;
  int ct   = rest / nsteps;
  int k    = st * 32 + ((lane & 16) >> 1) + (e & 7) + ((e & 8) << 1);
  int col  = ct * 16 + (lane & 15);
  P[idx] = (k < K) ? (__bf16)W[(size_t)k * DIMF + col] : (__bf16)0.0f;
}

// ---------------- bf16 WMMA GEMM: C[N,128] = A[N,K] * B[K,128] (+bias)(+relu) --------
// block = 256 threads = 8 waves; wave w -> cols [16w,16w+16); block covers 64 rows
// (4 row sub-tiles per wave, B fragment reused across the 4 accumulators).
template<int K>
__global__ void k_gemm(const float* __restrict__ A, const __bf16* __restrict__ Bp,
                       const float* __restrict__ bias, float* __restrict__ C,
                       int Nrows, int relu)
{
  constexpr int NSTEP = (K + 31) / 32;
  const int lane    = threadIdx.x & 31;
  const int wave    = threadIdx.x >> 5;
  const int rowBase = blockIdx.x * 64;
  const int col     = wave * 16 + (lane & 15);
  const int abase0  = (lane & 16) >> 1;     // A frag: lanes 0-15 -> K+0, lanes 16-31 -> K+8

  const float* Ap[4];
#pragma unroll
  for (int t = 0; t < 4; ++t){
    int r = rowBase + t * 16 + (lane & 15);
    if (r > Nrows - 1) r = Nrows - 1;       // clamp; stores are guarded below
    Ap[t] = A + (size_t)r * K;
  }
  const __bf16* bp = Bp + ((size_t)wave * NSTEP * 32 + lane) * 16;

  v8f acc[4] = {v8f{}, v8f{}, v8f{}, v8f{}};

  for (int st = 0; st < NSTEP; ++st){
    const int kb  = st * 32;
    const int kg0 = kb + abase0;            // elements 0..7  -> k = kg0 + e
    const int kg1 = kb + abase0 + 16;       // elements 8..15 -> k = kg1 + e
    const int   o0 = (kg0 + 8 <= K) ? kg0 : 0;   // branch-free tail handling
    const int   o1 = (kg1 + 8 <= K) ? kg1 : 0;
    const float m0 = (kg0 + 8 <= K) ? 1.0f : 0.0f;
    const float m1 = (kg1 + 8 <= K) ? 1.0f : 0.0f;

    const v16bf b = *(const v16bf*)(bp + (size_t)st * 32 * 16);

    v16bf a[4];
#pragma unroll
    for (int t = 0; t < 4; ++t){
      float4 f0 = ld4(Ap[t] + o0);
      float4 f1 = ld4(Ap[t] + o0 + 4);
      float4 f2 = ld4(Ap[t] + o1);
      float4 f3 = ld4(Ap[t] + o1 + 4);
      a[t][0]  = (__bf16)(f0.x * m0); a[t][1]  = (__bf16)(f0.y * m0);
      a[t][2]  = (__bf16)(f0.z * m0); a[t][3]  = (__bf16)(f0.w * m0);
      a[t][4]  = (__bf16)(f1.x * m0); a[t][5]  = (__bf16)(f1.y * m0);
      a[t][6]  = (__bf16)(f1.z * m0); a[t][7]  = (__bf16)(f1.w * m0);
      a[t][8]  = (__bf16)(f2.x * m1); a[t][9]  = (__bf16)(f2.y * m1);
      a[t][10] = (__bf16)(f2.z * m1); a[t][11] = (__bf16)(f2.w * m1);
      a[t][12] = (__bf16)(f3.x * m1); a[t][13] = (__bf16)(f3.y * m1);
      a[t][14] = (__bf16)(f3.z * m1); a[t][15] = (__bf16)(f3.w * m1);
    }
    acc[0] = __builtin_amdgcn_wmma_f32_16x16x32_bf16(false, a[0], false, b, (short)0, acc[0], false, false);
    acc[1] = __builtin_amdgcn_wmma_f32_16x16x32_bf16(false, a[1], false, b, (short)0, acc[1], false, false);
    acc[2] = __builtin_amdgcn_wmma_f32_16x16x32_bf16(false, a[2], false, b, (short)0, acc[2], false, false);
    acc[3] = __builtin_amdgcn_wmma_f32_16x16x32_bf16(false, a[3], false, b, (short)0, acc[3], false, false);
  }

  const float bv = bias ? bias[col] : 0.0f;
#pragma unroll
  for (int t = 0; t < 4; ++t){
    const int r0 = rowBase + t * 16 + ((lane & 16) >> 1);  // C/D: lanes 0-15 rows r, 16-31 rows r+8
#pragma unroll
    for (int r = 0; r < 8; ++r){
      int row = r0 + r;
      if (row < Nrows){
        float v = acc[t][r] + bv;
        if (relu) v = fmaxf(v, 0.0f);
        C[(size_t)row * DIMF + col] = v;
      }
    }
  }
}

// ---------------- BatchNorm over axis 0 ----------------
__global__ void k_bn_reduce(const float* __restrict__ h, float* bns, float* bnq, int N){
  __shared__ float ssum[256], ssq[256];
  int t = threadIdx.x;
  int c = t & 127, half = t >> 7;
  int chunk = (N + gridDim.x - 1) / gridDim.x;
  int r0 = blockIdx.x * chunk;
  int r1 = min(r0 + chunk, N);
  float s = 0.f, q = 0.f;
  for (int r = r0 + half; r < r1; r += 2){
    float v = h[(size_t)r * DIMF + c];
    s += v; q += v * v;
  }
  ssum[t] = s; ssq[t] = q;
  __syncthreads();
  if (t < 128){
    atomicAdd(&bns[c], ssum[t] + ssum[t + 128]);
    atomicAdd(&bnq[c], ssq[t]  + ssq[t + 128]);
  }
}

__global__ void k_bn_apply(float* __restrict__ h, const float* bns, const float* bnq,
                           const float* __restrict__ g, const float* __restrict__ b, int N){
  int idx = blockIdx.x * blockDim.x + threadIdx.x;
  if (idx >= N * DIMF) return;
  int c = idx & 127;
  float inv = 1.0f / (float)N;
  float mu  = bns[c] * inv;
  float var = bnq[c] * inv - mu * mu;
  float v = (h[idx] - mu) * rsqrtf(var + 1e-5f) * g[c] + b[c];
  h[idx] = fmaxf(v, 0.0f);
}

// ---------------- per-node attention scores ----------------
__global__ void k_scores(const float* __restrict__ hp, const float* __restrict__ as,
                         const float* __restrict__ ad, float* __restrict__ asrc,
                         float* __restrict__ adst, int N){
  int idx = blockIdx.x * blockDim.x + threadIdx.x;
  if (idx >= N * HDS) return;
  int n = idx >> 2, hh = idx & 3;
  const float* hr = hp + (size_t)n * DIMF + hh * CHN;
  const float* a1 = as + hh * CHN;
  const float* a2 = ad + hh * CHN;
  float sa = 0.f, sd = 0.f;
#pragma unroll
  for (int c = 0; c < CHN; ++c){ float v = hr[c]; sa += v * a1[c]; sd += v * a2[c]; }
  asrc[idx] = sa; adst[idx] = sd;
}

// ---------------- per-layer init: xatt = bias, mmax = -inf, denom = 0 ----------------
__global__ void k_init_att(float* __restrict__ xatt, float* __restrict__ mmax,
                           float* __restrict__ den, const float* __restrict__ gb, int N){
  int idx = blockIdx.x * blockDim.x + threadIdx.x;
  if (idx >= N * DIMF) return;
  int c = idx & 127;
  xatt[idx] = gb[c];
  if (c < HDS){
    int n = idx >> 7;
    mmax[n * HDS + c] = -INFINITY;
    den[n * HDS + c]  = 0.f;
  }
}

__device__ __forceinline__ void atomicMaxF(float* addr, float v){
  if (v >= 0.0f) atomicMax((int*)addr, __float_as_int(v));
  else           atomicMin((unsigned int*)addr, __float_as_uint(v));
}

__device__ __forceinline__ void edge_sd(const int* __restrict__ ei, int i, int E, int& s, int& d){
  if (i < E){ s = ei[i]; d = ei[E + i]; } else { s = i - E; d = s; }
}

// ---------------- edge passes: score+max, exp+sum, normalize ----------------
__global__ void k_edge_scores(const int* __restrict__ ei, const float* __restrict__ asrc,
                              const float* __restrict__ adst, float* __restrict__ att,
                              float* __restrict__ mmax, int E, int Etot){
  int idx = blockIdx.x * blockDim.x + threadIdx.x;
  if (idx >= Etot * HDS) return;
  int i = idx >> 2, hh = idx & 3, s, d;
  edge_sd(ei, i, E, s, d);
  float e = asrc[s * HDS + hh] + adst[d * HDS + hh];
  e = (e > 0.f) ? e : 0.2f * e;
  att[idx] = e;
  atomicMaxF(&mmax[d * HDS + hh], e);
}

__global__ void k_edge_exp(const int* __restrict__ ei, float* __restrict__ att,
                           const float* __restrict__ mmax, float* __restrict__ den,
                           int E, int Etot){
  int idx = blockIdx.x * blockDim.x + threadIdx.x;
  if (idx >= Etot * HDS) return;
  int i = idx >> 2, hh = idx & 3, s, d;
  edge_sd(ei, i, E, s, d);
  float ex = expf(att[idx] - mmax[d * HDS + hh]);
  att[idx] = ex;
  atomicAdd(&den[d * HDS + hh], ex);
}

__global__ void k_edge_alpha(const int* __restrict__ ei, float* __restrict__ att,
                             const float* __restrict__ den, int E, int Etot){
  int idx = blockIdx.x * blockDim.x + threadIdx.x;
  if (idx >= Etot * HDS) return;
  int i = idx >> 2, hh = idx & 3, s, d;
  edge_sd(ei, i, E, s, d);
  att[idx] = att[idx] / (den[d * HDS + hh] + 1e-16f);
}

// ---------------- message passing: xatt[dst] += hp[src] * alpha ----------------
__global__ void k_message(const int* __restrict__ ei, const float* __restrict__ hp,
                          const float* __restrict__ att, float* __restrict__ xatt,
                          int E, int Etot){
  long long idx = (long long)blockIdx.x * blockDim.x + threadIdx.x;
  if (idx >= (long long)Etot * 32) return;
  int i  = (int)(idx >> 5);
  int c0 = ((int)(idx & 31)) * 4;
  int s, d;
  edge_sd(ei, i, E, s, d);
  float al = att[i * HDS + (c0 >> 5)];
  float4 hv = ld4(hp + (size_t)s * DIMF + c0);
  float* o = xatt + (size_t)d * DIMF + c0;
  atomicAdd(o + 0, hv.x * al);
  atomicAdd(o + 1, hv.y * al);
  atomicAdd(o + 2, hv.z * al);
  atomicAdd(o + 3, hv.w * al);
}

// ---------------- LayerNorm(x + res), in place into x; one wave per row ----------------
__global__ void k_ln(float* __restrict__ x, const float* __restrict__ res,
                     const float* __restrict__ g, const float* __restrict__ b, int N){
  int wid  = (blockIdx.x * blockDim.x + threadIdx.x) >> 5;
  int lane = threadIdx.x & 31;
  if (wid >= N) return;
  float* row = x + (size_t)wid * DIMF;
  float4 v = ld4(row + lane * 4);
  float4 r = ld4(res + (size_t)wid * DIMF + lane * 4);
  v.x += r.x; v.y += r.y; v.z += r.z; v.w += r.w;
  float s  = v.x + v.y + v.z + v.w;
  float sq = v.x*v.x + v.y*v.y + v.z*v.z + v.w*v.w;
#pragma unroll
  for (int m = 16; m > 0; m >>= 1){ s += __shfl_xor(s, m); sq += __shfl_xor(sq, m); }
  float mean = s * (1.0f / DIMF);
  float var  = sq * (1.0f / DIMF) - mean * mean;
  float rstd = rsqrtf(var + 1e-5f);
  int c = lane * 4;
  row[c + 0] = (v.x - mean) * rstd * g[c + 0] + b[c + 0];
  row[c + 1] = (v.y - mean) * rstd * g[c + 1] + b[c + 1];
  row[c + 2] = (v.z - mean) * rstd * g[c + 2] + b[c + 2];
  row[c + 3] = (v.w - mean) * rstd * g[c + 3] + b[c + 3];
}

// ---------------- classifier ----------------
__global__ void k_cls(const float* __restrict__ h, const float* __restrict__ w,
                      const float* __restrict__ b, float* __restrict__ out, int N){
  int n = blockIdx.x * blockDim.x + threadIdx.x;
  if (n >= N) return;
  const float4* hr = (const float4*)(h + (size_t)n * DIMF);
  const float4* wr = (const float4*)w;
  float acc = b[0];
#pragma unroll
  for (int j = 0; j < DIMF / 4; ++j){
    float4 a = hr[j], ww = wr[j];
    acc += a.x*ww.x + a.y*ww.y + a.z*ww.z + a.w*ww.w;
  }
  out[n] = acc;
}

extern "C" void kernel_launch(void* const* d_in, const int* in_sizes, int n_in,
                              void* d_out, int out_size, void* d_ws, size_t ws_size,
                              hipStream_t stream)
{
  const float* x      = (const float*)d_in[0];
  const int*   ei     = (const int*)d_in[1];
  const float* emb_w  = (const float*)d_in[2];
  const float* emb_b  = (const float*)d_in[3];
  const float* bn_g   = (const float*)d_in[4];
  const float* bn_b   = (const float*)d_in[5];
  const float* gat_w  = (const float*)d_in[6];
  const float* att_s  = (const float*)d_in[7];
  const float* att_d  = (const float*)d_in[8];
  const float* gat_b  = (const float*)d_in[9];
  const float* ln1_g  = (const float*)d_in[10];
  const float* ln1_b  = (const float*)d_in[11];
  const float* ffn_w1 = (const float*)d_in[12];
  const float* ffn_b1 = (const float*)d_in[13];
  const float* ffn_w2 = (const float*)d_in[14];
  const float* ffn_b2 = (const float*)d_in[15];
  const float* ln2_g  = (const float*)d_in[16];
  const float* ln2_b  = (const float*)d_in[17];
  const float* cls_w  = (const float*)d_in[18];
  const float* cls_b  = (const float*)d_in[19];

  const int N    = in_sizes[0] / 600;   // 50000
  const int E    = in_sizes[1] / 2;     // 800000
  const int Etot = E + N;               // 850000
  const int L    = 3;

  float* out = (float*)d_out;

  // workspace layout (floats)
  float* h    = (float*)d_ws;
  float* hp   = h    + (size_t)N * DIMF;
  float* xatt = hp   + (size_t)N * DIMF;
  float* asrc = xatt + (size_t)N * DIMF;
  float* adst = asrc + (size_t)N * HDS;
  float* mmax = adst + (size_t)N * HDS;
  float* den  = mmax + (size_t)N * HDS;
  float* bns  = den  + (size_t)N * HDS;
  float* bnq  = bns  + DIMF;
  float* bsum = bnq  + DIMF;
  // packed bf16 weights (64B aligned region after the float slabs)
  size_t fOff = (size_t)(bsum + DIMF - (float*)d_ws);
  fOff = (fOff + 15) & ~(size_t)15;     // 64-byte align
  __bf16* pEmb = (__bf16*)((float*)d_ws + fOff);
  const int PACK600 = 8 * ((600 + 31) / 32) * 32 * 16;   // 77824
  const int PACK128 = 8 * ((128 + 31) / 32) * 32 * 16;   // 16384
  __bf16* pW[3][3];                     // [layer][gat, ffn1, ffn2]
  {
    __bf16* p = pEmb + PACK600;
    for (int l = 0; l < 3; ++l)
      for (int j = 0; j < 3; ++j){ pW[l][j] = p; p += PACK128; }
  }

  dim3 b256(256);
  const int rowTiles64 = (N + 63) / 64;
  const int gNodeC     = (N * DIMF + 255) / 256;
  const int gNodeH     = (N * HDS + 255) / 256;
  const int gEdgeH     = (Etot * HDS + 255) / 256;
  const int gMsg       = (Etot * 32 + 255) / 256;
  const int gLN        = (N * 32 + 255) / 256;

  // pack all weights (tiny; L2-resident thereafter)
  k_packB<<<(PACK600 + 255) / 256, b256, 0, stream>>>(emb_w, pEmb, 600);
  for (int l = 0; l < L; ++l){
    k_packB<<<(PACK128 + 255) / 256, b256, 0, stream>>>(gat_w  + (size_t)l * DIMF * DIMF, pW[l][0], DIMF);
    k_packB<<<(PACK128 + 255) / 256, b256, 0, stream>>>(ffn_w1 + (size_t)l * DIMF * DIMF, pW[l][1], DIMF);
    k_packB<<<(PACK128 + 255) / 256, b256, 0, stream>>>(ffn_w2 + (size_t)l * DIMF * DIMF, pW[l][2], DIMF);
  }

  // embedding + batchnorm + relu
  k_prep<<<1, 128, 0, stream>>>(emb_b, bsum, bns, bnq);
  k_gemm<600><<<rowTiles64, b256, 0, stream>>>(x, pEmb, bsum, h, N, 0);
  k_bn_reduce<<<200, b256, 0, stream>>>(h, bns, bnq, N);
  k_bn_apply<<<gNodeC, b256, 0, stream>>>(h, bns, bnq, bn_g, bn_b, N);

  for (int l = 0; l < L; ++l){
    float* att = out + N + (size_t)l * Etot * HDS;   // attention output slab (also softmax scratch)
    k_gemm<128><<<rowTiles64, b256, 0, stream>>>(h, pW[l][0], nullptr, hp, N, 0);
    k_scores<<<gNodeH, b256, 0, stream>>>(hp, att_s + l * HDS * CHN, att_d + l * HDS * CHN, asrc, adst, N);
    k_init_att<<<gNodeC, b256, 0, stream>>>(xatt, mmax, den, gat_b + l * DIMF, N);
    k_edge_scores<<<gEdgeH, b256, 0, stream>>>(ei, asrc, adst, att, mmax, E, Etot);
    k_edge_exp<<<gEdgeH, b256, 0, stream>>>(ei, att, mmax, den, E, Etot);
    k_edge_alpha<<<gEdgeH, b256, 0, stream>>>(ei, att, den, E, Etot);
    k_message<<<gMsg, b256, 0, stream>>>(ei, hp, att, xatt, E, Etot);
    k_ln<<<gLN, b256, 0, stream>>>(h, xatt, ln1_g + l * DIMF, ln1_b + l * DIMF, N);
    k_gemm<128><<<rowTiles64, b256, 0, stream>>>(h, pW[l][1], ffn_b1 + l * DIMF, hp, N, 1);
    k_gemm<128><<<rowTiles64, b256, 0, stream>>>(hp, pW[l][2], ffn_b2 + l * DIMF, xatt, N, 0);
    k_ln<<<gLN, b256, 0, stream>>>(h, xatt, ln2_g + l * DIMF, ln2_b + l * DIMF, N);
  }
  k_cls<<<(N + 255) / 256, b256, 0, stream>>>(h, cls_w, cls_b, out, N);
}
